// DynamicDecoder_53558242181341
// MI455X (gfx1250) — matrix operand order repaired and checked by reading, not compiled, not run on hardware
//
#include <hip/hip_runtime.h>

// ---------------- problem constants ----------------
#define Hc   200
#define Pc   4
#define Tc   4
#define Bc   64
#define Mc   1000
#define NEGV (-1e30f)

constexpr int K1  = 600;   // 3H   (GEMM1 K)
constexpr int K1P = 608;   // padded to 32
constexpr int KC1 = K1P / 32;   // 19
constexpr int K2  = 200;   // H    (GEMM2 K)
constexpr int K2P = 224;
constexpr int KC2 = K2P / 32;   // 7
constexpr int N1  = 800;   // H*P
constexpr int NT  = N1 / 16;    // 50 real n-tiles
constexpr int NTP = 56;         // padded n-tiles (8 waves x 7), extra tiles are zero weights
constexpr int MT  = (Mc + 15) / 16; // 63 m-tiles

typedef __attribute__((ext_vector_type(16))) __bf16 v16bf;
typedef __attribute__((ext_vector_type(8)))  float  v8f;

__device__ inline unsigned short f2bf(float f) {
  union { float f; unsigned int u; } x; x.f = f;
  unsigned int r = x.u + 0x7fffu + ((x.u >> 16) & 1u);
  return (unsigned short)(r >> 16);
}

__device__ inline v16bf mkfrag(uint4 a, uint4 b) {
  union { uint4 u[2]; v16bf v; } x;
  x.u[0] = a; x.u[1] = b;
  return x.v;
}

// ---------------- fp32 -> bf16 bulk convert ----------------
__global__ void k_cvt(const float* __restrict__ src, unsigned short* __restrict__ dst, long long n) {
  long long i = (long long)blockIdx.x * blockDim.x + threadIdx.x;
  long long stride = (long long)gridDim.x * blockDim.x;
  for (; i < n; i += stride) dst[i] = f2bf(src[i]);
}

// ---------------- repack W (N x K, row-major f32) into WMMA B-fragment layout ----------------
// packed[(((nt*KC + kc)*32 + lane)*16 + i] = bf16( W[n, k] ),
//   n = nt*16 + (lane&15), k = kc*32 + (lane>>4)*16 + i  (zero if k>=Kdim or n>=Ndim)
__global__ void k_pack(const float* __restrict__ W, unsigned short* __restrict__ pW,
                       int Kdim, int KC, int Ndim) {
  int nt = blockIdx.x, kc = blockIdx.y, lane = threadIdx.x;
  int n = nt * 16 + (lane & 15);
  int kbase = kc * 32 + (lane >> 4) * 16;
  unsigned short* dst = pW + ((((size_t)nt * KC + kc) * 32 + lane) << 4);
  for (int i = 0; i < 16; ++i) {
    int k = kbase + i;
    dst[i] = (k < Kdim && n < Ndim) ? f2bf(W[(size_t)n * Kdim + k]) : (unsigned short)0;
  }
}

// ---------------- init: h=c=0, s=0, e=sum(mask)-1, u_s=U[b,0], masks=1 ----------------
__global__ void k_init(const float* __restrict__ U, const int* __restrict__ d_mask,
                       float* h, float* c, float* us,
                       int* sidx, int* eidx, int* ms, int* me, float* lossvec) {
  int b = blockIdx.x, tid = threadIdx.x;
  __shared__ int red[256];
  int cnt = 0;
  for (int m = tid; m < Mc; m += 256) cnt += d_mask[b * Mc + m];
  red[tid] = cnt; __syncthreads();
  for (int s = 128; s > 0; s >>= 1) { if (tid < s) red[tid] += red[tid + s]; __syncthreads(); }
  if (tid == 0) {
    eidx[b] = red[0] - 1; sidx[b] = 0; ms[b] = 1; me[b] = 1; lossvec[b] = 0.f;
  }
  for (int j = tid; j < 400; j += 256) us[b * 400 + j] = U[(size_t)b * Mc * 400 + j];
  for (int j = tid; j < Hc;  j += 256) { h[b * Hc + j] = 0.f; c[b * Hc + j] = 0.f; }
}

// ---------------- LSTM step (small, VALU) ----------------
__global__ void k_lstm(const float* __restrict__ U,
                       const float* __restrict__ w_ih, const float* __restrict__ w_hh,
                       const float* __restrict__ b_ih, const float* __restrict__ b_hh,
                       const float* __restrict__ us, const int* __restrict__ eidx,
                       float* h, float* c, float* ucat) {
  int b = blockIdx.x, tid = threadIdx.x;
  __shared__ float x[800];
  __shared__ float hs[Hc];
  __shared__ float g[800];
  int e = eidx[b];
  const float* ue = U + (size_t)b * Mc * 400 + (size_t)e * 400;
  for (int j = tid; j < 400; j += 256) { x[j] = us[b * 400 + j]; x[400 + j] = ue[j]; }
  for (int j = tid; j < Hc; j += 256) hs[j] = h[b * Hc + j];
  __syncthreads();
  for (int j = tid; j < 800; j += 256) {
    float acc = b_ih[j] + b_hh[j];
    const float* wi = w_ih + (size_t)j * 800;
    for (int k = 0; k < 800; ++k) acc += x[k] * wi[k];
    const float* wh = w_hh + (size_t)j * Hc;
    for (int k = 0; k < Hc; ++k) acc += hs[k] * wh[k];
    g[j] = acc;
  }
  __syncthreads();
  for (int j = tid; j < Hc; j += 256) {
    float i_ = 1.f / (1.f + __expf(-g[j]));
    float f_ = 1.f / (1.f + __expf(-g[Hc + j]));
    float gg = tanhf(g[2 * Hc + j]);
    float o_ = 1.f / (1.f + __expf(-g[3 * Hc + j]));
    float cn = f_ * c[b * Hc + j] + i_ * gg;
    c[b * Hc + j] = cn;
    h[b * Hc + j] = o_ * tanhf(cn);
  }
  for (int j = tid; j < 800; j += 256) ucat[b * 800 + j] = x[j];
}

// ---------------- r = tanh([h, u_cat] @ Wr^T), bf16 out ----------------
__global__ void k_r(const float* __restrict__ h, const float* __restrict__ ucat,
                    const float* __restrict__ Wr, unsigned short* __restrict__ rbf) {
  int b = blockIdx.x, tid = threadIdx.x;
  __shared__ float cat[1000];
  for (int j = tid; j < Hc; j += 256) cat[j] = h[b * Hc + j];
  for (int j = tid; j < 800; j += 256) cat[Hc + j] = ucat[b * 800 + j];
  __syncthreads();
  for (int j = tid; j < Hc; j += 256) {
    const float* w = Wr + (size_t)j * 1000;
    float acc = 0.f;
    for (int k = 0; k < 1000; ++k) acc += cat[k] * w[k];
    rbf[b * Hc + j] = f2bf(tanhf(acc));
  }
}

// ---------------- fused HMN over a 16-row M tile (WMMA) ----------------
__global__ __launch_bounds__(256, 1)
void k_hmn(const unsigned short* __restrict__ Ubf,
           const unsigned short* __restrict__ rbf,
           const unsigned short* __restrict__ pW1,
           const unsigned short* __restrict__ pW2,
           const float* __restrict__ b1, const float* __restrict__ b2,
           const float* __restrict__ W12, const float* __restrict__ b12,
           const int* __restrict__ d_mask,
           float* __restrict__ alpha) {
  __shared__ __align__(16) unsigned short xs[16 * K1P];   // x tile: [U(400)|r(200)|pad(8)]
  __shared__ __align__(16) unsigned short m1b[16 * K2P];  // m1 bf16 (GEMM2 A), K padded
  __shared__ float m1f[16 * K2];
  __shared__ float m2f[16 * K2];
  __shared__ float ctile[8][16][16];                      // per-wave C staging
  __shared__ float ap[16][4];

  int tid = threadIdx.x;
  int wv = tid >> 5, lane = tid & 31;
  int b  = blockIdx.y;
  int m0 = blockIdx.x * 16;

  // ---- stage x tile into LDS ----
  for (int i = tid; i < 16 * K1P; i += 256) {
    int row = i / K1P, col = i - row * K1P;
    int m = m0 + row;
    unsigned short v = 0;
    if (m < Mc) {
      if (col < 400)      v = Ubf[((size_t)b * Mc + m) * 400 + col];
      else if (col < 600) v = rbf[b * Hc + (col - 400)];
    }
    xs[i] = v;
  }
  for (int i = tid; i < 16 * K2P; i += 256) m1b[i] = 0;
  __syncthreads();

  int row   = lane & 15;
  int koffA = (lane >> 4) * 16;
  v8f zero = {};

  // ---- GEMM1: x(16x608) @ W1^T -> 16x800 ----
  // wave wv owns n-tiles [wv*7, wv*7+7), all valid thanks to NTP=56 zero padding:
  // straight-line (no EXEC-divergent control flow around WMMA), loads clause ahead of MMAs.
  v8f acc[7];
#pragma unroll
  for (int i = 0; i < 7; ++i) acc[i] = zero;
  {
    const unsigned short* bb = pW1 + (((size_t)(wv * 7) * KC1) * 32 + lane) * 16;
    constexpr size_t tstride = (size_t)KC1 * 32 * 16;  // elements between n-tiles
    for (int kc = 0; kc < KC1; ++kc) {
      const unsigned short* abase = &xs[row * K1P + kc * 32 + koffA];
      v16bf afrag = mkfrag(*(const uint4*)abase, *(const uint4*)(abase + 8));
      const unsigned short* bk = bb + (size_t)kc * 32 * 16;
      if (kc + 1 < KC1) __builtin_prefetch(bk + 32 * 16, 0, 1);
      uint4 blo[7], bhi[7];
#pragma unroll
      for (int i = 0; i < 7; ++i) {
        const uint4* pb = (const uint4*)(bk + (size_t)i * tstride);
        blo[i] = pb[0]; bhi[i] = pb[1];
      }
#pragma unroll
      for (int i = 0; i < 7; ++i) {
        acc[i] = __builtin_amdgcn_wmma_f32_16x16x32_bf16(
            false, afrag, false, mkfrag(blo[i], bhi[i]), (short)0, acc[i], false, false);
      }
    }
  }
  // ---- maxout (P=4 along N) + b1 -> m1 ----
#pragma unroll
  for (int i = 0; i < 7; ++i) {
    int nt = wv * 7 + i;
    if (nt < NT) {
      int rbase = (lane >> 4) * 8;
#pragma unroll
      for (int r = 0; r < 8; ++r) ctile[wv][r + rbase][lane & 15] = acc[i][r];
      for (int q = lane; q < 64; q += 32) {
        int rr = q & 15, j = q >> 4;
        float mx = -3.4e38f;
#pragma unroll
        for (int p = 0; p < 4; ++p)
          mx = fmaxf(mx, ctile[wv][rr][4 * j + p] + b1[nt * 16 + 4 * j + p]);
        int cc = nt * 4 + j;
        m1f[rr * K2 + cc] = mx;
        m1b[rr * K2P + cc] = f2bf(mx);
      }
    }
  }
  __syncthreads();

  // ---- GEMM2: m1(16x224) @ W2^T -> 16x800 ----
  v8f acc2[7];
#pragma unroll
  for (int i = 0; i < 7; ++i) acc2[i] = zero;
  {
    const unsigned short* bb = pW2 + (((size_t)(wv * 7) * KC2) * 32 + lane) * 16;
    constexpr size_t tstride = (size_t)KC2 * 32 * 16;
    for (int kc = 0; kc < KC2; ++kc) {
      const unsigned short* abase = &m1b[row * K2P + kc * 32 + koffA];
      v16bf afrag = mkfrag(*(const uint4*)abase, *(const uint4*)(abase + 8));
      const unsigned short* bk = bb + (size_t)kc * 32 * 16;
      uint4 blo[7], bhi[7];
#pragma unroll
      for (int i = 0; i < 7; ++i) {
        const uint4* pb = (const uint4*)(bk + (size_t)i * tstride);
        blo[i] = pb[0]; bhi[i] = pb[1];
      }
#pragma unroll
      for (int i = 0; i < 7; ++i) {
        acc2[i] = __builtin_amdgcn_wmma_f32_16x16x32_bf16(
            false, afrag, false, mkfrag(blo[i], bhi[i]), (short)0, acc2[i], false, false);
      }
    }
  }
  // ---- maxout + b2 -> m2 ----
#pragma unroll
  for (int i = 0; i < 7; ++i) {
    int nt = wv * 7 + i;
    if (nt < NT) {
      int rbase = (lane >> 4) * 8;
#pragma unroll
      for (int r = 0; r < 8; ++r) ctile[wv][r + rbase][lane & 15] = acc2[i][r];
      for (int q = lane; q < 64; q += 32) {
        int rr = q & 15, j = q >> 4;
        float mx = -3.4e38f;
#pragma unroll
        for (int p = 0; p < 4; ++p)
          mx = fmaxf(mx, ctile[wv][rr][4 * j + p] + b2[nt * 16 + 4 * j + p]);
        m2f[rr * K2 + nt * 4 + j] = mx;
      }
    }
  }
  __syncthreads();

  // ---- alpha = max_p([m1,m2] @ W12^T + b12), apply mask ----
  if (tid < 64) {
    int rr = tid >> 2, p = tid & 3;
    const float* w = W12 + (size_t)p * 400;
    float a = b12[p];
    for (int k = 0; k < K2; ++k) a += m1f[rr * K2 + k] * w[k];
    for (int k = 0; k < K2; ++k) a += m2f[rr * K2 + k] * w[200 + k];
    ap[rr][p] = a;
  }
  __syncthreads();
  if (tid < 16) {
    int m = m0 + tid;
    if (m < Mc) {
      float a = fmaxf(fmaxf(ap[tid][0], ap[tid][1]), fmaxf(ap[tid][2], ap[tid][3]));
      float mk = (float)d_mask[b * Mc + m];
      alpha[(size_t)b * Mc + m] = a + (1.f - mk) * NEGV;
    }
  }
}

// ---------------- per-batch argmax + log-softmax value at target ----------------
__global__ void k_reduce(const float* __restrict__ alpha, const int* __restrict__ span,
                         int which, int* __restrict__ oidx, float* __restrict__ oval) {
  int b = blockIdx.x, tid = threadIdx.x;
  __shared__ float smax[256];
  __shared__ int   simx[256];
  __shared__ float ssum[256];
  const float* a = alpha + (size_t)b * Mc;
  float mx = -3.4e38f; int mi = 0;
  for (int m = tid; m < Mc; m += 256) { float v = a[m]; if (v > mx) { mx = v; mi = m; } }
  smax[tid] = mx; simx[tid] = mi; __syncthreads();
  for (int s = 128; s > 0; s >>= 1) {
    if (tid < s) {
      if (smax[tid + s] > smax[tid] ||
          (smax[tid + s] == smax[tid] && simx[tid + s] < simx[tid])) {
        smax[tid] = smax[tid + s]; simx[tid] = simx[tid + s];
      }
    }
    __syncthreads();
  }
  float gmax = smax[0];
  float se = 0.f;
  for (int m = tid; m < Mc; m += 256) se += __expf(a[m] - gmax);
  ssum[tid] = se; __syncthreads();
  for (int s = 128; s > 0; s >>= 1) { if (tid < s) ssum[tid] += ssum[tid + s]; __syncthreads(); }
  if (tid == 0) {
    oidx[b] = simx[0];
    int tgt = span[b * 2 + which];
    oval[b] = a[tgt] - gmax - __logf(ssum[0]);
  }
}

// ---------------- sequential control: update idx/mask/loss, optional u_s gather ----------------
__global__ void k_control(const float* __restrict__ U,
                          const int* __restrict__ oidx, const float* __restrict__ oval,
                          int* cur, int* msk, int* hist_idx, int* hist_msk,
                          float* lossvec, float* us,
                          const float* __restrict__ ucat, float* ucat2,
                          int t, int do_gather) {
  int tid = threadIdx.x;
  __shared__ float vred[64];
  __shared__ int snew[64];
  if (tid < 64) vred[tid] = oval[tid];
  __syncthreads();
  for (int s = 32; s > 0; s >>= 1) { if (tid < s) vred[tid] += vred[tid + s]; __syncthreads(); }
  float meanv = vred[0] / 64.f;
  if (tid < 64) {
    int b = tid;
    int idx = oidx[b];
    int old = cur[b], om = msk[b];
    int nv = (t == 0) ? idx : idx * om;
    int nm = (t == 0) ? 1 : ((nv != old * om) ? 1 : 0);
    cur[b] = nv; msk[b] = nm;
    hist_idx[t * 64 + b] = nv; hist_msk[t * 64 + b] = nm;
    lossvec[b] += -meanv * (float)nm;
    snew[b] = nv;
  }
  __syncthreads();
  if (do_gather) {
    for (int i = tid; i < 64 * 400; i += 256) {
      int b = i / 400, j = i - b * 400;
      float v = U[((size_t)b * Mc + snew[b]) * 400 + j];
      us[b * 400 + j] = v;
      ucat2[b * 800 + j] = v;
      ucat2[b * 800 + 400 + j] = ucat[b * 800 + 400 + j];
    }
  }
}

// ---------------- finalize: p1, p2, loss ----------------
__global__ void k_finalize(const int* __restrict__ ss, const int* __restrict__ mss,
                           const int* __restrict__ es, const int* __restrict__ mes,
                           const float* __restrict__ lossvec, float* __restrict__ out) {
  int tid = threadIdx.x;
  __shared__ float lred[64];
  if (tid < 64) {
    int b = tid;
    int ps = 0, pe = 0;
    for (int t = 0; t < Tc; ++t) { ps += mss[t * 64 + b]; pe += mes[t * 64 + b]; }
    out[1 + b]      = (float)ss[(ps - 1) * 64 + b];
    out[1 + 64 + b] = (float)es[(pe - 1) * 64 + b];
    lred[tid] = lossvec[b];
  }
  __syncthreads();
  for (int s = 32; s > 0; s >>= 1) { if (tid < s) lred[tid] += lred[tid + s]; __syncthreads(); }
  if (tid == 0) out[0] = lred[0] / 64.f / (float)Tc;
}

// ---------------- host ----------------
extern "C" void kernel_launch(void* const* d_in, const int* in_sizes, int n_in,
                              void* d_out, int out_size, void* d_ws, size_t ws_size,
                              hipStream_t stream) {
  (void)in_sizes; (void)n_in; (void)out_size; (void)ws_size;
  const float* U     = (const float*)d_in[0];
  const int*   dmask = (const int*)d_in[1];
  const int*   span  = (const int*)d_in[2];
  const float* w_ih  = (const float*)d_in[3];
  const float* w_hh  = (const float*)d_in[4];
  const float* b_ih  = (const float*)d_in[5];
  const float* b_hh  = (const float*)d_in[6];
  const float* Wr[2]  = { (const float*)d_in[7],  (const float*)d_in[14] };
  const float* W1[2]  = { (const float*)d_in[8],  (const float*)d_in[15] };
  const float* b1[2]  = { (const float*)d_in[9],  (const float*)d_in[16] };
  const float* W2[2]  = { (const float*)d_in[10], (const float*)d_in[17] };
  const float* b2[2]  = { (const float*)d_in[11], (const float*)d_in[18] };
  const float* W12[2] = { (const float*)d_in[12], (const float*)d_in[19] };
  const float* b12[2] = { (const float*)d_in[13], (const float*)d_in[20] };

  char* ws = (char*)d_ws;
  size_t off = 0;
  auto alloc = [&](size_t bytes) -> void* {
    void* p = ws + off;
    off = (off + bytes + 255) & ~(size_t)255;
    return p;
  };
  unsigned short* Ubf = (unsigned short*)alloc((size_t)Bc * Mc * 400 * 2);
  unsigned short* pW1b[2], * pW2b[2];
  for (int tg = 0; tg < 2; ++tg) {
    pW1b[tg] = (unsigned short*)alloc((size_t)NTP * KC1 * 32 * 16 * 2);
    pW2b[tg] = (unsigned short*)alloc((size_t)NTP * KC2 * 32 * 16 * 2);
  }
  float* alpha   = (float*)alloc((size_t)Bc * Mc * 4);
  unsigned short* rbf = (unsigned short*)alloc((size_t)Bc * Hc * 2);
  float* h       = (float*)alloc((size_t)Bc * Hc * 4);
  float* c       = (float*)alloc((size_t)Bc * Hc * 4);
  float* ucat    = (float*)alloc((size_t)Bc * 800 * 4);
  float* ucat2   = (float*)alloc((size_t)Bc * 800 * 4);
  float* us      = (float*)alloc((size_t)Bc * 400 * 4);
  int* sidx      = (int*)alloc(Bc * 4);
  int* eidx      = (int*)alloc(Bc * 4);
  int* ms        = (int*)alloc(Bc * 4);
  int* me        = (int*)alloc(Bc * 4);
  int* ss        = (int*)alloc(Tc * Bc * 4);
  int* mss       = (int*)alloc(Tc * Bc * 4);
  int* es        = (int*)alloc(Tc * Bc * 4);
  int* mes       = (int*)alloc(Tc * Bc * 4);
  int* ridx      = (int*)alloc(Bc * 4);
  float* rval    = (float*)alloc(Bc * 4);
  float* lossvec = (float*)alloc(Bc * 4);

  // one-time per launch: bf16 U + packed (zero-padded) weights
  k_cvt<<<8192, 256, 0, stream>>>(U, Ubf, (long long)Bc * Mc * 400);
  for (int tg = 0; tg < 2; ++tg) {
    k_pack<<<dim3(NTP, KC1), 32, 0, stream>>>(W1[tg], pW1b[tg], K1, KC1, N1);
    k_pack<<<dim3(NTP, KC2), 32, 0, stream>>>(W2[tg], pW2b[tg], K2, KC2, N1);
  }
  k_init<<<Bc, 256, 0, stream>>>(U, dmask, h, c, us, sidx, eidx, ms, me, lossvec);

  for (int t = 0; t < Tc; ++t) {
    k_lstm<<<Bc, 256, 0, stream>>>(U, w_ih, w_hh, b_ih, b_hh, us, eidx, h, c, ucat);

    // ---- start head ----
    k_r<<<Bc, 256, 0, stream>>>(h, ucat, Wr[0], rbf);
    k_hmn<<<dim3(MT, Bc), 256, 0, stream>>>(Ubf, rbf, pW1b[0], pW2b[0],
                                            b1[0], b2[0], W12[0], b12[0], dmask, alpha);
    k_reduce<<<Bc, 256, 0, stream>>>(alpha, span, 0, ridx, rval);
    k_control<<<1, 256, 0, stream>>>(U, ridx, rval, sidx, ms, ss, mss,
                                     lossvec, us, ucat, ucat2, t, /*gather=*/1);

    // ---- end head ----
    k_r<<<Bc, 256, 0, stream>>>(h, ucat2, Wr[1], rbf);
    k_hmn<<<dim3(MT, Bc), 256, 0, stream>>>(Ubf, rbf, pW1b[1], pW2b[1],
                                            b1[1], b2[1], W12[1], b12[1], dmask, alpha);
    k_reduce<<<Bc, 256, 0, stream>>>(alpha, span, 1, ridx, rval);
    k_control<<<1, 256, 0, stream>>>(U, ridx, rval, eidx, me, es, mes,
                                     lossvec, us, ucat, ucat2, t, /*gather=*/0);
  }

  k_finalize<<<1, 64, 0, stream>>>(ss, mss, es, mes, lossvec, (float*)d_out);
}
